// Bert_Encoder_38439957299768
// MI455X (gfx1250) — matrix operand order repaired and checked by reading, not compiled, not run on hardware
//
#include <hip/hip_runtime.h>
#include <hip/hip_bf16.h>

// ---------------------------------------------------------------------------
// Types for CDNA5 WMMA (wave32): v16bf A/B operands, v8f accumulator
// ---------------------------------------------------------------------------
typedef __bf16 bf16_t;
typedef __attribute__((ext_vector_type(16))) __bf16 v16bf;
typedef __attribute__((ext_vector_type(8)))  __bf16 v8bf;
typedef __attribute__((ext_vector_type(8)))  float  v8f;
typedef __attribute__((ext_vector_type(4)))  int    v4i;

#define AS1 __attribute__((address_space(1)))
#define AS3 __attribute__((address_space(3)))

#if defined(__AMDGCN__) && __has_builtin(__builtin_amdgcn_global_load_async_to_lds_b128)
#define HAVE_ASYNC_LDS 1
#else
#define HAVE_ASYNC_LDS 0
#endif

#define HD   768
#define FD   3072
#define NHEAD 12
#define DH   64
#define SEQ  2048
#define NTOK 4096   // 2 * 2048

__device__ __forceinline__ __bf16 f2bf(float f) {
    unsigned u = __builtin_bit_cast(unsigned, f);
    unsigned r = (u + 0x7FFFu + ((u >> 16) & 1u)) >> 16;   // RNE
    unsigned short h = (unsigned short)r;
    return __builtin_bit_cast(__bf16, h);
}

__device__ __forceinline__ v8f v8f_zero() {
    v8f z;
#pragma unroll
    for (int i = 0; i < 8; i++) z[i] = 0.0f;
    return z;
}

// A-operand (16x32 bf16) load, row-major source, ld = row stride (elements).
// lane<16: row=l, K = k+{0..15}; lane>=16: row=l, K = k+{16..31}
__device__ __forceinline__ v16bf load_a16(const __bf16* __restrict__ X, int ld,
                                          int row, int k, int lane) {
    int half = lane >> 4;
    const __bf16* p0 = X + (size_t)row * ld + k + half * 16;
    v8bf lo = *(const v8bf*)p0;
    v8bf hi = *(const v8bf*)(p0 + 8);
    v16bf r;
#pragma unroll
    for (int i = 0; i < 8; i++) { r[i] = lo[i]; r[i + 8] = hi[i]; }
    return r;
}

// B-operand (32x16 bf16) load from K-contiguous (transposed) source:
// Wt[col][k .. k+15 (+16 for upper half)] is contiguous -> single 32B load.
__device__ __forceinline__ v16bf load_b16(const __bf16* __restrict__ Wt, int ldk,
                                          int col, int k, int half) {
    const __bf16* p = Wt + (size_t)col * ldk + k + half * 16;
    return *(const v16bf*)p;
}

// B-operand read from LDS-staged tile (column stride 40 halfs = 80B,
// conflict-free: (col*20 + j) mod 64 distinct for 16 cols).
__device__ __forceinline__ v16bf load_b_lds(const __bf16* buf, int col, int half) {
    const __bf16* p = buf + col * 40 + half * 16;
    v8bf lo = *(const v8bf*)p;
    v8bf hi = *(const v8bf*)(p + 8);
    v16bf r;
#pragma unroll
    for (int i = 0; i < 8; i++) { r[i] = lo[i]; r[i + 8] = hi[i]; }
    return r;
}

__device__ __forceinline__ v8f wmma_bf16(v16bf a, v16bf b, v8f c) {
    return __builtin_amdgcn_wmma_f32_16x16x32_bf16(false, a, false, b,
                                                   (short)0, c, false, false);
}

// ---------------------------------------------------------------------------
// Conversion / transpose kernels
// ---------------------------------------------------------------------------
__global__ void cvt_f32_to_bf16_kernel(const float* __restrict__ src,
                                       __bf16* __restrict__ dst, int n) {
    int i = blockIdx.x * 256 + threadIdx.x;
    if (i < n) dst[i] = f2bf(src[i]);
}

// W[K][Ncol] f32 -> Wt[Ncol][K] bf16
__global__ void wtrans_kernel(const float* __restrict__ W, __bf16* __restrict__ Wt,
                              int K, int Ncol) {
    int i = blockIdx.x * 256 + threadIdx.x;
    if (i < K * Ncol) {
        int n = i / K, kk = i - n * K;
        Wt[i] = f2bf(W[(size_t)kk * Ncol + n]);
    }
}

// v[token][h*64+d] bf16 -> vT[((b*12+h)*64+d)][s] bf16
__global__ void vtrans_kernel(const __bf16* __restrict__ v, __bf16* __restrict__ vT) {
    int i = blockIdx.x * 256 + threadIdx.x;
    if (i < NTOK * HD) {
        int s = i & (SEQ - 1);
        int rest = i >> 11;
        int d = rest & (DH - 1);
        int bh = rest >> 6;
        int b = bh / NHEAD, h = bh - b * NHEAD;
        vT[i] = v[(size_t)(b * SEQ + s) * HD + h * DH + d];
    }
}

// ---------------------------------------------------------------------------
// Generic bf16 WMMA GEMM: Y = X[M,K] @ Wt[N,K]^T + bias
// EPI: 0 = store f32, 1 = store bf16, 2 = exact GELU then bf16
// Block: 128 threads (4 waves). Wave tile 32x64, block tile 128x64.
// B-tile (shared by all 4 waves) is staged in LDS with double-buffered
// GLOBAL_LOAD_ASYNC_TO_LDS_B128 when the toolchain exposes the builtin.
// ---------------------------------------------------------------------------
template <int EPI>
__global__ void __launch_bounds__(128)
gemm_bf16_kernel(const __bf16* __restrict__ X, const __bf16* __restrict__ Wt,
                 const float* __restrict__ bias, float* __restrict__ Yf,
                 __bf16* __restrict__ Yb, int M, int Ncol, int K) {
#if HAVE_ASYNC_LDS
    __shared__ __bf16 LdsB[2][64 * 40];  // 2 x 5120B double buffer
#endif
    int lane = threadIdx.x & 31;
    int wave = threadIdx.x >> 5;
    int l = lane & 15, half = lane >> 4;
    int rowbase = blockIdx.y * 128 + wave * 32;
    int cbase   = blockIdx.x * 64;

    v8f acc[2][4];
#pragma unroll
    for (int i = 0; i < 2; i++)
#pragma unroll
        for (int j = 0; j < 4; j++) acc[i][j] = v8f_zero();

#if HAVE_ASYNC_LDS
    // 64 cols x 32 k-values (64B/col) -> 256 16B segments, 2 per thread.
    auto issueB = [&](int k, int buf) {
#pragma unroll
        for (int i = 0; i < 2; i++) {
            int seg = threadIdx.x + i * 128;
            int c = seg >> 2, s = seg & 3;
            const __bf16* gp = Wt + (size_t)(cbase + c) * K + k + s * 8;
            __bf16* lp = &LdsB[buf][c * 40 + s * 8];
            __builtin_amdgcn_global_load_async_to_lds_b128(
                (AS1 v4i*)gp, (AS3 v4i*)lp, 0, 0);
        }
    };
    int cur = 0;
    issueB(0, 0);
    for (int k = 0; k < K; k += 32) {
        asm volatile("s_wait_asynccnt 0" ::: "memory");
        __syncthreads();                 // buf[cur] filled, prev reads done
        if (k + 32 < K) issueB(k + 32, cur ^ 1);
        v16bf a0 = load_a16(X, K, rowbase + l, k, lane);
        v16bf a1 = load_a16(X, K, rowbase + 16 + l, k, lane);
        const __bf16* bt = &LdsB[cur][0];
#pragma unroll
        for (int nb = 0; nb < 4; nb++) {
            v16bf b = load_b_lds(bt, nb * 16 + l, half);
            acc[0][nb] = wmma_bf16(a0, b, acc[0][nb]);
            acc[1][nb] = wmma_bf16(a1, b, acc[1][nb]);
        }
        cur ^= 1;
    }
#else
    for (int k = 0; k < K; k += 32) {
        v16bf a0 = load_a16(X, K, rowbase + l, k, lane);
        v16bf a1 = load_a16(X, K, rowbase + 16 + l, k, lane);
#pragma unroll
        for (int nb = 0; nb < 4; nb++) {
            v16bf b = load_b16(Wt, K, cbase + nb * 16 + l, k, half);
            acc[0][nb] = wmma_bf16(a0, b, acc[0][nb]);
            acc[1][nb] = wmma_bf16(a1, b, acc[1][nb]);
        }
    }
#endif

#pragma unroll
    for (int hrow = 0; hrow < 2; hrow++) {
#pragma unroll
        for (int nb = 0; nb < 4; nb++) {
            int col = cbase + nb * 16 + l;
            float bv = bias[col];
#pragma unroll
            for (int r = 0; r < 8; r++) {
                int row = rowbase + hrow * 16 + half * 8 + r;
                float vv = acc[hrow][nb][r] + bv;
                if (EPI == 2) vv = 0.5f * vv * (1.0f + erff(vv * 0.70710678118654752f));
                size_t o = (size_t)row * Ncol + col;
                if (EPI == 0) Yf[o] = vv;
                else          Yb[o] = f2bf(vv);
            }
        }
    }
}

// ---------------------------------------------------------------------------
// Flash attention: one wave = one (b, h, 16-query tile). 32 keys per step.
// q,k: bf16 [NTOK, HD]; vT: bf16 [b,h,d,s]; ctx out: bf16 [NTOK, HD]
// ---------------------------------------------------------------------------
__global__ void __launch_bounds__(128)
attn_kernel(const __bf16* __restrict__ q, const __bf16* __restrict__ kmat,
            const __bf16* __restrict__ vT, __bf16* __restrict__ ctx) {
    __shared__ __bf16 Plds[4][16][40];   // per-wave P staging, 80B row stride
    int lane = threadIdx.x & 31;
    int wave = threadIdx.x >> 5;
    int l = lane & 15, half = lane >> 4;

    int gw = blockIdx.x * 4 + wave;      // 0 .. 3071
    int qt = gw & 127;                   // 128 query tiles
    int bh = gw >> 7;                    // 0 .. 23
    int h = bh % NHEAD, b = bh / NHEAD;
    int q0 = qt * 16;

    const __bf16* kHead  = kmat + (size_t)b * SEQ * HD + h * DH;
    const __bf16* vTHead = vT + (size_t)bh * DH * SEQ;

    v16bf qa0 = load_a16(q, HD, b * SEQ + q0 + l, h * DH, lane);
    v16bf qa1 = load_a16(q, HD, b * SEQ + q0 + l, h * DH + 32, lane);

    v8f ctxa[4];
#pragma unroll
    for (int g = 0; g < 4; g++) ctxa[g] = v8f_zero();
    float m[8], lsum[8];
#pragma unroll
    for (int r = 0; r < 8; r++) { m[r] = -1e30f; lsum[r] = 0.0f; }

    for (int kn = 0; kn < SEQ; kn += 32) {
        v8f s0 = v8f_zero(), s1 = v8f_zero();
        s0 = wmma_bf16(qa0, load_b16(kHead, HD, kn + l, 0, half), s0);
        s0 = wmma_bf16(qa1, load_b16(kHead, HD, kn + l, 32, half), s0);
        s1 = wmma_bf16(qa0, load_b16(kHead, HD, kn + 16 + l, 0, half), s1);
        s1 = wmma_bf16(qa1, load_b16(kHead, HD, kn + 16 + l, 32, half), s1);

        float alpha[8];
#pragma unroll
        for (int r = 0; r < 8; r++) {
            float a = s0[r] * 0.125f;      // 1/sqrt(64)
            float c = s1[r] * 0.125f;
            float rm = fmaxf(a, c);
            rm = fmaxf(rm, __shfl_xor(rm, 8));
            rm = fmaxf(rm, __shfl_xor(rm, 4));
            rm = fmaxf(rm, __shfl_xor(rm, 2));
            rm = fmaxf(rm, __shfl_xor(rm, 1));   // row max within 16-lane half
            float mn = fmaxf(m[r], rm);
            alpha[r] = __expf(m[r] - mn);
            m[r] = mn;
            float p0 = __expf(a - mn), p1 = __expf(c - mn);
            float rs = p0 + p1;
            rs += __shfl_xor(rs, 8);
            rs += __shfl_xor(rs, 4);
            rs += __shfl_xor(rs, 2);
            rs += __shfl_xor(rs, 1);
            lsum[r] = lsum[r] * alpha[r] + rs;
            // C-layout -> LDS (row = query, col = key offset)
            Plds[wave][half * 8 + r][l]      = f2bf(p0);
            Plds[wave][half * 8 + r][l + 16] = f2bf(p1);
        }
#pragma unroll
        for (int g = 0; g < 4; g++)
#pragma unroll
            for (int r = 0; r < 8; r++) ctxa[g][r] *= alpha[r];

        // same-wave LDS ops are in-order; wait + memory clobber stops reordering
        asm volatile("s_wait_dscnt 0" ::: "memory");

        // Re-read P in A-matrix layout
        const __bf16* pw = &Plds[wave][0][0];
        int rowoff = l * 40 + half * 16;
        v8bf plo = *(const v8bf*)(pw + rowoff);
        v8bf phi = *(const v8bf*)(pw + rowoff + 8);
        v16bf pa;
#pragma unroll
        for (int i = 0; i < 8; i++) { pa[i] = plo[i]; pa[i + 8] = phi[i]; }

#pragma unroll
        for (int g = 0; g < 4; g++)
            ctxa[g] = wmma_bf16(pa, load_b16(vTHead, SEQ, g * 16 + l, kn, half), ctxa[g]);
    }

    float inv[8];
#pragma unroll
    for (int r = 0; r < 8; r++) inv[r] = 1.0f / lsum[r];
#pragma unroll
    for (int g = 0; g < 4; g++)
#pragma unroll
        for (int r = 0; r < 8; r++) {
            int row = b * SEQ + q0 + half * 8 + r;
            ctx[(size_t)row * HD + h * DH + g * 16 + l] = f2bf(ctxa[g][r] * inv[r]);
        }
}

// ---------------------------------------------------------------------------
// LayerNorm(Y)*gamma+beta + residual; writes f32 (+ optional bf16 copy)
// One block (256 threads) per 768-wide row.
// ---------------------------------------------------------------------------
__global__ void __launch_bounds__(256)
ln_res_kernel(const float* __restrict__ Y, const float* __restrict__ gamma,
              const float* __restrict__ beta, const float* __restrict__ resid,
              float* __restrict__ outF, __bf16* __restrict__ outB) {
    __shared__ float red[16];
    int row = blockIdx.x;
    int tid = threadIdx.x;
    int lane = tid & 31, wave = tid >> 5;
    const float* y = Y + (size_t)row * HD;
    float v0 = y[tid], v1 = y[tid + 256], v2 = y[tid + 512];
    float s = v0 + v1 + v2;
    float ss = v0 * v0 + v1 * v1 + v2 * v2;
#pragma unroll
    for (int off = 16; off >= 1; off >>= 1) {
        s  += __shfl_xor(s, off);
        ss += __shfl_xor(ss, off);
    }
    if (lane == 0) { red[wave] = s; red[wave + 8] = ss; }
    __syncthreads();
    float ts = 0.0f, tss = 0.0f;
#pragma unroll
    for (int w = 0; w < 8; w++) { ts += red[w]; tss += red[w + 8]; }
    float mu = ts * (1.0f / HD);
    float var = tss * (1.0f / HD) - mu * mu;
    float rstd = rsqrtf(var + 1e-5f);
    const float* r = resid + (size_t)row * HD;
#pragma unroll
    for (int j = 0; j < 3; j++) {
        int i = tid + j * 256;
        float o = (y[i] - mu) * rstd * gamma[i] + beta[i] + r[i];
        outF[(size_t)row * HD + i] = o;
        if (outB) outB[(size_t)row * HD + i] = f2bf(o);
    }
}

// ---------------------------------------------------------------------------
// Host launch
// ---------------------------------------------------------------------------
extern "C" void kernel_launch(void* const* d_in, const int* in_sizes, int n_in,
                              void* d_out, int out_size, void* d_ws, size_t ws_size,
                              hipStream_t stream) {
    const float* x   = (const float*)d_in[0];
    const float* Wq  = (const float*)d_in[1];  const float* bq  = (const float*)d_in[2];
    const float* Wk  = (const float*)d_in[3];  const float* bk  = (const float*)d_in[4];
    const float* Wv  = (const float*)d_in[5];  const float* bv  = (const float*)d_in[6];
    const float* Wa1 = (const float*)d_in[7];  const float* ba1 = (const float*)d_in[8];
    const float* g1  = (const float*)d_in[9];  const float* be1 = (const float*)d_in[10];
    const float* W1  = (const float*)d_in[11]; const float* b1  = (const float*)d_in[12];
    const float* W2  = (const float*)d_in[13]; const float* b2  = (const float*)d_in[14];
    const float* Wa2 = (const float*)d_in[15]; const float* ba2 = (const float*)d_in[16];
    const float* g2  = (const float*)d_in[17]; const float* be2 = (const float*)d_in[18];

    char* ws = (char*)d_ws;
    size_t off = 0;
    auto alloc = [&](size_t bytes) -> void* {
        off = (off + 255) & ~(size_t)255;
        void* p = ws + off;
        off += bytes;
        return p;
    };

    __bf16* xb    = (__bf16*)alloc((size_t)NTOK * HD * 2);
    __bf16* Wqt   = (__bf16*)alloc((size_t)HD * HD * 2);
    __bf16* Wkt   = (__bf16*)alloc((size_t)HD * HD * 2);
    __bf16* Wvt   = (__bf16*)alloc((size_t)HD * HD * 2);
    __bf16* Wa1t  = (__bf16*)alloc((size_t)HD * HD * 2);
    __bf16* Wa2t  = (__bf16*)alloc((size_t)HD * HD * 2);
    __bf16* W1t   = (__bf16*)alloc((size_t)HD * FD * 2);
    __bf16* W2t   = (__bf16*)alloc((size_t)HD * FD * 2);
    __bf16* qb    = (__bf16*)alloc((size_t)NTOK * HD * 2);
    __bf16* kb    = (__bf16*)alloc((size_t)NTOK * HD * 2);
    __bf16* vb    = (__bf16*)alloc((size_t)NTOK * HD * 2);
    __bf16* vTb   = (__bf16*)alloc((size_t)NTOK * HD * 2);
    __bf16* ctxb  = (__bf16*)alloc((size_t)NTOK * HD * 2);
    float*  a1f   = (float*)alloc((size_t)NTOK * HD * 4);
    float*  an1f  = (float*)alloc((size_t)NTOK * HD * 4);
    __bf16* an1b  = (__bf16*)alloc((size_t)NTOK * HD * 2);
    __bf16* h1b   = (__bf16*)alloc((size_t)NTOK * FD * 2);
    __bf16* h2b   = (__bf16*)alloc((size_t)NTOK * HD * 2);
    float*  a2f   = (float*)alloc((size_t)NTOK * HD * 4);

    // --- convert inputs to bf16 (weights transposed to [Ncol][K]) ---
    {
        int n = NTOK * HD;
        cvt_f32_to_bf16_kernel<<<(n + 255) / 256, 256, 0, stream>>>(x, xb, n);
        int nw = HD * HD;
        wtrans_kernel<<<(nw + 255) / 256, 256, 0, stream>>>(Wq,  Wqt,  HD, HD);
        wtrans_kernel<<<(nw + 255) / 256, 256, 0, stream>>>(Wk,  Wkt,  HD, HD);
        wtrans_kernel<<<(nw + 255) / 256, 256, 0, stream>>>(Wv,  Wvt,  HD, HD);
        wtrans_kernel<<<(nw + 255) / 256, 256, 0, stream>>>(Wa1, Wa1t, HD, HD);
        wtrans_kernel<<<(nw + 255) / 256, 256, 0, stream>>>(Wa2, Wa2t, HD, HD);
        int nf = HD * FD;
        wtrans_kernel<<<(nf + 255) / 256, 256, 0, stream>>>(W1, W1t, HD, FD);   // -> [FD][HD]
        wtrans_kernel<<<(nf + 255) / 256, 256, 0, stream>>>(W2, W2t, FD, HD);   // -> [HD][FD]
    }

    dim3 gH(HD / 64, NTOK / 128);   // 12 x 32
    dim3 gF(FD / 64, NTOK / 128);   // 48 x 32

    // --- QKV projections ---
    gemm_bf16_kernel<1><<<gH, 128, 0, stream>>>(xb, Wqt, bq, nullptr, qb, NTOK, HD, HD);
    gemm_bf16_kernel<1><<<gH, 128, 0, stream>>>(xb, Wkt, bk, nullptr, kb, NTOK, HD, HD);
    gemm_bf16_kernel<1><<<gH, 128, 0, stream>>>(xb, Wvt, bv, nullptr, vb, NTOK, HD, HD);
    {
        int n = NTOK * HD;
        vtrans_kernel<<<(n + 255) / 256, 256, 0, stream>>>(vb, vTb);
    }

    // --- attention: 2*12*128 waves, 4 waves per block ---
    attn_kernel<<<768, 128, 0, stream>>>(qb, kb, vTb, ctxb);

    // --- add_norm_1: ctx @ Wa1 + ba1 -> LN -> + x ---
    gemm_bf16_kernel<0><<<gH, 128, 0, stream>>>(ctxb, Wa1t, ba1, a1f, nullptr, NTOK, HD, HD);
    ln_res_kernel<<<NTOK, 256, 0, stream>>>(a1f, g1, be1, x, an1f, an1b);

    // --- FFN: GELU(an1 @ W1 + b1) @ W2 + b2 ---
    gemm_bf16_kernel<2><<<gF, 128, 0, stream>>>(an1b, W1t, b1, nullptr, h1b, NTOK, FD, HD);
    gemm_bf16_kernel<1><<<gH, 128, 0, stream>>>(h1b, W2t, b2, nullptr, h2b, NTOK, HD, FD);

    // --- add_norm_2: h2 @ Wa2 + ba2 -> LN -> + an1 ---
    gemm_bf16_kernel<0><<<gH, 128, 0, stream>>>(h2b, Wa2t, ba2, a2f, nullptr, NTOK, HD, HD);
    ln_res_kernel<<<NTOK, 256, 0, stream>>>(a2f, g2, be2, an1f, (float*)d_out, nullptr);
}
